// YOLO1_53317724012852
// MI455X (gfx1250) — matrix operand order repaired and checked by reading, not compiled, not run on hardware
//
#include <hip/hip_runtime.h>
#include <cmath>

typedef __attribute__((ext_vector_type(16))) _Float16 v16h;
typedef __attribute__((ext_vector_type(8)))  _Float16 v8h;
typedef __attribute__((ext_vector_type(8)))  float    v8f;

#define BATCH      4
#define IMGSZ      448
#define NCLS       20
#define FEATW      14
#define NCELL      (FEATW*FEATW)
#define COUT_FINAL 30

// CDNA5 async global->LDS path (ASYNCcnt) — probe-verified on this toolchain.
#if defined(__HIP_DEVICE_COMPILE__) && \
    __has_builtin(__builtin_amdgcn_global_load_async_to_lds_b128) && \
    __has_builtin(__builtin_amdgcn_s_wait_asynccnt)
#define USE_ASYNC_LDS 1
typedef int v4i_a __attribute__((vector_size(16)));
typedef __attribute__((address_space(1))) v4i_a* async_gptr;
typedef __attribute__((address_space(3))) v4i_a* async_lptr;
#else
#define USE_ASYNC_LDS 0
#endif

// ---------------------------------------------------------------------------
// fp32 -> fp16 conversion (image, layer-0 weights, head weights)
// ---------------------------------------------------------------------------
__global__ void f32_to_f16_kernel(const float* __restrict__ src,
                                  _Float16* __restrict__ dst, int n) {
    int i = blockIdx.x * blockDim.x + threadIdx.x;
    if (i < n) dst[i] = (_Float16)src[i];
}

// ---------------------------------------------------------------------------
// Pack conv weights OIHW(f32) -> [tap][cout][cin](f16): A-tile rows contiguous
// ---------------------------------------------------------------------------
__global__ void pack_w_kernel(const float* __restrict__ w,
                              _Float16* __restrict__ wt, int Cout, int Cin) {
    int total = Cout * Cin * 9;
    int i = blockIdx.x * blockDim.x + threadIdx.x;
    if (i >= total) return;
    int ci  = i % Cin;
    int co  = (i / Cin) % Cout;
    int tap = i / (Cin * Cout);
    wt[i] = (_Float16)w[(co * Cin + ci) * 9 + tap];
}

// ---------------------------------------------------------------------------
// Layer 1: Cin=3 direct conv (27-term reduction, memory bound -> VALU)
// ---------------------------------------------------------------------------
__global__ void conv3x3_c3_kernel(const _Float16* __restrict__ x,
                                  const _Float16* __restrict__ wf,
                                  const float* __restrict__ bias,
                                  _Float16* __restrict__ y) {
    const int H = IMGSZ, W = IMGSZ, HW = H * W;
    int idx = blockIdx.x * blockDim.x + threadIdx.x;
    if (idx >= BATCH * 64 * HW) return;
    int pos = idx % HW;
    int co  = (idx / HW) % 64;
    int b   = idx / (HW * 64);
    int h = pos / W, w = pos % W;
    float s = bias[co];
    #pragma unroll
    for (int ci = 0; ci < 3; ++ci)
        #pragma unroll
        for (int kh = 0; kh < 3; ++kh) {
            int hh = h + kh - 1;
            if (hh < 0 || hh >= H) continue;
            #pragma unroll
            for (int kw = 0; kw < 3; ++kw) {
                int ww = w + kw - 1;
                if (ww < 0 || ww >= W) continue;
                s += (float)x[(b * 3 + ci) * HW + hh * W + ww] *
                     (float)wf[((co * 3 + ci) * 3 + kh) * 3 + kw];
            }
        }
    y[idx] = (_Float16)(s > 0.f ? s : 0.f);
}

// ---------------------------------------------------------------------------
// Implicit-GEMM conv3x3 + bias + ReLU via v_wmma_f32_16x16x32_f16.
// Templated on layer shape: K strides become instruction immediates, so the
// 16 B-staging loads clause with zero per-load VALU.
// Block = 256 thr = 8 waves (2 m x 4 n), wave tile 2x2 frags -> 4 WMMA/K-step.
// Block tile 64 cout x 128 pixels; A weights via async global->LDS b128.
// ---------------------------------------------------------------------------
#define AST  40   // A LDS row stride (halves): 80B rows, 16B-aligned
#define BSTP 40   // B LDS row stride (halves): 80B rows, 16B-aligned

template <int CIN, int COUT, int H, int W>
__global__ __launch_bounds__(256)
void conv3x3_wmma_t(const _Float16* __restrict__ x,
                    const _Float16* __restrict__ wt,   // [tap][co][ci] f16
                    const float* __restrict__ bias,
                    _Float16* __restrict__ y) {
    constexpr int HW = H * W;
    __shared__ _Float16 At[64 * AST];      // 64 cout x 32 k
    __shared__ _Float16 Bt[128 * BSTP];    // 128 pix x 32 k  ([n][k])
    __shared__ int hwq[128];               // packed (h<<16)|w per tile pixel

    const int t     = threadIdx.x;
    const int lane  = t & 31;
    const int wv    = t >> 5;
    const int mwave = wv & 1;              // 2 cout half-tiles
    const int nwave = wv >> 1;             // 4 pixel quarter-tiles
    const int pos0  = blockIdx.x * 128;
    const int co0   = blockIdx.y * 64;
    const int b     = blockIdx.z;
    const _Float16* xb = x + (size_t)b * CIN * HW;

    if (t < 128) {
        int pos = pos0 + t;
        hwq[t] = (pos < HW) ? (((pos / W) << 16) | (pos % W)) : -1;
    }
    __syncthreads();

    const int am = t >> 1;                 // A stage: row handled by thread
    const int ak = (t & 1) * 16;           // A stage: 16-half column base
    const int nB = t & 127;                // B stage: pixel owned by thread
    const int kqB = t >> 7;                // B stage: K half (0: k0-15, 1: k16-31)
    const int hwn = hwq[nB];
    const int mrow  = mwave * 32 + (lane & 15);
    const int koffA = (lane >> 4) * 8;
    const int nrow  = nwave * 32 + (lane & 15);
    const int koffB = (lane >> 4) * 16;

    v8f acc[2][2];
    #pragma unroll
    for (int i = 0; i < 2; ++i)
        #pragma unroll
        for (int j = 0; j < 2; ++j)
            acc[i][j] = (v8f){0.f,0.f,0.f,0.f,0.f,0.f,0.f,0.f};

    #pragma unroll 1
    for (int tap = 0; tap < 9; ++tap) {
        const int kh = tap / 3 - 1;
        const int kw = tap % 3 - 1;
        const _Float16* wtap = wt + (size_t)tap * COUT * CIN;

        // per-tap boundary resolution for this thread's pixel (hoisted out of K)
        bool bval = false;
        int  boff = 0;
        if (hwn >= 0) {
            int hh = (hwn >> 16) + kh;
            int ww = (hwn & 0xffff) + kw;
            if (((unsigned)hh < (unsigned)H) & ((unsigned)ww < (unsigned)W)) {
                bval = true;
                boff = hh * W + ww;
            }
        }

        #pragma unroll 1
        for (int ck = 0; ck < CIN; ck += 32) {
            // ---- stage A (weights): async 16B per thread -> LDS ----
#if USE_ASYNC_LDS
            if (t < 128)
                __builtin_amdgcn_global_load_async_to_lds_b128(
                    (async_gptr)(wtap + (co0 + am) * CIN + ck + ak),
                    (async_lptr)&At[am * AST + ak], 0, 0);
#else
            if (t < 128) {
                const _Float16* g = wtap + (co0 + am) * CIN + ck + ak;
                #pragma unroll
                for (int q = 0; q < 16; ++q) At[am * AST + ak + q] = g[q];
            }
#endif
            // speculative prefetch of the next activation chunk
            __builtin_prefetch(xb + (ck + 32 < CIN ? ck + 32 : 0) * HW + pos0, 0, 1);

            // ---- stage B (activations): 16-wide K run, immediate offsets ----
            v8h t0 = {(_Float16)0,(_Float16)0,(_Float16)0,(_Float16)0,
                      (_Float16)0,(_Float16)0,(_Float16)0,(_Float16)0};
            v8h t1 = t0;
            if (bval) {
                const _Float16* src = xb + (size_t)(ck + kqB * 16) * HW + boff;
                #pragma unroll
                for (int j = 0; j < 8; ++j) t0[j] = src[(size_t)j * HW];
                #pragma unroll
                for (int j = 0; j < 8; ++j) t1[j] = src[(size_t)(j + 8) * HW];
            }
            *(v8h*)&Bt[nB * BSTP + kqB * 16]     = t0;
            *(v8h*)&Bt[nB * BSTP + kqB * 16 + 8] = t1;
#if USE_ASYNC_LDS
            __builtin_amdgcn_s_wait_asynccnt(0);
#endif
            __syncthreads();

            // ---- fragment loads (contiguous k = 16*(lane>>4) + s for A and B; b128 merged) ----
            union U { v16h v; unsigned u[8]; };
            U A0, A1, B0, B1;
            #pragma unroll
            for (int p = 0; p < 8; ++p) {
                int j = 2 * p;
                int kk = j + koffB;
                A0.u[p] = *(const unsigned*)&At[mrow * AST + kk];
                A1.u[p] = *(const unsigned*)&At[(mrow + 16) * AST + kk];
                B0.u[p] = *(const unsigned*)&Bt[nrow * BSTP + koffB + 2 * p];
                B1.u[p] = *(const unsigned*)&Bt[(nrow + 16) * BSTP + koffB + 2 * p];
            }
            __syncthreads();

            acc[0][0] = __builtin_amdgcn_wmma_f32_16x16x32_f16(
                            false, A0.v, false, B0.v, (short)0, acc[0][0], false, false);
            acc[0][1] = __builtin_amdgcn_wmma_f32_16x16x32_f16(
                            false, A0.v, false, B1.v, (short)0, acc[0][1], false, false);
            acc[1][0] = __builtin_amdgcn_wmma_f32_16x16x32_f16(
                            false, A1.v, false, B0.v, (short)0, acc[1][0], false, false);
            acc[1][1] = __builtin_amdgcn_wmma_f32_16x16x32_f16(
                            false, A1.v, false, B1.v, (short)0, acc[1][1], false, false);
        }
    }

    // ---- epilogue: bias + ReLU, f16 store (C/D: M = r + 8*(lane>=16)) ----
    #pragma unroll
    for (int mf = 0; mf < 2; ++mf)
        #pragma unroll
        for (int nf = 0; nf < 2; ++nf) {
            int pos = pos0 + nwave * 32 + nf * 16 + (lane & 15);
            if (pos < HW) {
                #pragma unroll
                for (int r = 0; r < 8; ++r) {
                    int co = co0 + mwave * 32 + mf * 16 + r + 8 * (lane >> 4);
                    float v = acc[mf][nf][r] + bias[co];
                    y[(b * COUT + co) * HW + pos] = (_Float16)(v > 0.f ? v : 0.f);
                }
            }
        }
}

// ---------------------------------------------------------------------------
// 2x2 max pool, f16
// ---------------------------------------------------------------------------
__global__ void maxpool2x2_kernel(const _Float16* __restrict__ x,
                                  _Float16* __restrict__ y,
                                  int C, int H, int W) {
    int Ho = H / 2, Wo = W / 2;
    int total = BATCH * C * Ho * Wo;
    int idx = blockIdx.x * blockDim.x + threadIdx.x;
    if (idx >= total) return;
    int wo = idx % Wo;
    int ho = (idx / Wo) % Ho;
    int c  = (idx / (Wo * Ho)) % C;
    int b  = idx / (Wo * Ho * C);
    const _Float16* p = x + ((b * C + c) * H + ho * 2) * W + wo * 2;
    float m = (float)p[0];
    float v = (float)p[1];          m = v > m ? v : m;
    v = (float)p[W];                m = v > m ? v : m;
    v = (float)p[W + 1];            m = v > m ? v : m;
    y[idx] = (_Float16)m;
}

// ---------------------------------------------------------------------------
// Final 1x1 conv head (Cin=512 -> Cout=30), WMMA, fp32 output, no ReLU.
// Loads are unconditional with clamped M/N addresses — garbage rows/cols land
// only in D elements that the store mask already discards.
// ---------------------------------------------------------------------------
__global__ void conv1x1_wmma_kernel(const _Float16* __restrict__ x,
                                    const _Float16* __restrict__ wf,
                                    const float* __restrict__ bias,
                                    float* __restrict__ feat) {
    const int lane = threadIdx.x & 31;
    const int pos0 = blockIdx.x * 16;
    const int co0  = blockIdx.y * 16;
    const int b    = blockIdx.z;
    const int m    = co0 + (lane & 15);
    const int mc   = m < COUT_FINAL ? m : COUT_FINAL - 1;   // clamp, mask at store
    const int koffA = (lane >> 4) * 8;
    const int n    = pos0 + (lane & 15);
    const int nc   = n < NCELL ? n : NCELL - 1;             // clamp, mask at store
    const int koffB = (lane >> 4) * 16;
    const _Float16* wr = wf + mc * 512;
    const _Float16* xr = x + (size_t)b * 512 * NCELL + nc;

    v8f acc = {0.f, 0.f, 0.f, 0.f, 0.f, 0.f, 0.f, 0.f};
    for (int ck = 0; ck < 512; ck += 32) {
        union { v16h v; unsigned u[8]; } A;
        #pragma unroll
        for (int p = 0; p < 8; ++p) {
            int j = 2 * p;
            int k = j + koffB + ck;
            A.u[p] = *(const unsigned*)&wr[k];
        }
        union { v16h v; _Float16 h[16]; } B;
        #pragma unroll
        for (int j = 0; j < 16; ++j)
            B.h[j] = xr[(size_t)(ck + koffB + j) * NCELL];
        acc = __builtin_amdgcn_wmma_f32_16x16x32_f16(
                  false, A.v, false, B.v, (short)0, acc, false, false);
    }
    #pragma unroll
    for (int r = 0; r < 8; ++r) {
        int co  = co0 + r + 8 * (lane >> 4);
        int pos = pos0 + (lane & 15);
        if (co < COUT_FINAL && pos < NCELL)
            feat[(b * COUT_FINAL + co) * NCELL + pos] = acc[r] + bias[co];
    }
}

// ---------------------------------------------------------------------------
// Decode: sigmoid boxes, raw-logit confidence * softmax classes, argmax, xyxy.
// out layout (B, 196, 7): x1,y1,x2,y2,prob,cat,keep(filled by NMS)
// ---------------------------------------------------------------------------
__global__ void evaluate_kernel(const float* __restrict__ feat,
                                float* __restrict__ out) {
    int b = blockIdx.x;
    int cell = threadIdx.x;
    if (cell >= NCELL) return;
    float f[COUT_FINAL];
    #pragma unroll
    for (int c = 0; c < COUT_FINAL; ++c)
        f[c] = feat[(b * COUT_FINAL + c) * NCELL + cell];

    float con0 = f[4], con1 = f[9];
    int box_idx = (con1 > con0) ? 1 : 0;
    float con_max = con1 > con0 ? con1 : con0;

    float mx = f[10];
    #pragma unroll
    for (int c = 1; c < NCLS; ++c) mx = f[10 + c] > mx ? f[10 + c] : mx;
    float e[NCLS], s = 0.f;
    #pragma unroll
    for (int c = 0; c < NCLS; ++c) { e[c] = expf(f[10 + c] - mx); s += e[c]; }
    float inv = 1.f / s;
    float best = -INFINITY; int cat = 0;
    #pragma unroll
    for (int c = 0; c < NCLS; ++c) {
        float p = con_max * (e[c] * inv);      // con_max may be negative
        if (p > best) { best = p; cat = c; }
    }

    int g = box_idx * 5;
    float bx = 1.f / (1.f + expf(-f[g + 0]));
    float by = 1.f / (1.f + expf(-f[g + 1]));
    float bw = 1.f / (1.f + expf(-f[g + 2]));
    float bh = 1.f / (1.f + expf(-f[g + 3]));
    float gx = (float)(cell % FEATW);
    float gy = (float)(cell / FEATW);
    float cx = (bx + gx) * ((float)IMGSZ / FEATW);
    float cy = (by + gy) * ((float)IMGSZ / FEATW);
    float ww = bw * (float)IMGSZ;
    float hh = bh * (float)IMGSZ;

    float* o = out + (b * NCELL + cell) * 7;
    o[0] = cx - ww * 0.5f;
    o[1] = cy - hh * 0.5f;
    o[2] = cx + ww * 0.5f;
    o[3] = cy + hh * 0.5f;
    o[4] = best;
    o[5] = (float)cat;
}

// ---------------------------------------------------------------------------
// Per-image NMS (N=196, thr 0.8): stable desc sort + triangular suppression.
// ---------------------------------------------------------------------------
__device__ inline float iou_xyxy(const float* a, const float* bx) {
    float areaA = (a[2] - a[0]) * (a[3] - a[1]);
    float areaB = (bx[2] - bx[0]) * (bx[3] - bx[1]);
    float xx1 = fmaxf(a[0], bx[0]), yy1 = fmaxf(a[1], bx[1]);
    float xx2 = fminf(a[2], bx[2]), yy2 = fminf(a[3], bx[3]);
    float inter = fmaxf(xx2 - xx1, 0.f) * fmaxf(yy2 - yy1, 0.f);
    return inter / (areaA + areaB - inter + 1e-9f);
}

__global__ void nms_kernel(float* __restrict__ out) {
    __shared__ float box[NCELL * 4];
    __shared__ float sc[NCELL];
    __shared__ int   ct[NCELL];
    __shared__ int   ord[NCELL];
    __shared__ unsigned char used[NCELL];
    __shared__ unsigned char keep[NCELL];
    int b = blockIdx.x;
    for (int i = threadIdx.x; i < NCELL; i += blockDim.x) {
        const float* o = out + (b * NCELL + i) * 7;
        box[i * 4 + 0] = o[0]; box[i * 4 + 1] = o[1];
        box[i * 4 + 2] = o[2]; box[i * 4 + 3] = o[3];
        sc[i] = o[4]; ct[i] = (int)o[5];
        used[i] = 0;
    }
    __syncthreads();
    if (threadIdx.x == 0) {
        for (int r = 0; r < NCELL; ++r) {
            int bi = -1; float bv = 0.f;
            for (int i = 0; i < NCELL; ++i)
                if (!used[i] && (bi < 0 || sc[i] > bv)) { bi = i; bv = sc[i]; }
            ord[r] = bi; used[bi] = 1; keep[r] = 1;
        }
        for (int i = 0; i < NCELL; ++i) {
            if (!keep[i]) continue;
            const float* bi = &box[ord[i] * 4];
            int ci = ct[ord[i]];
            for (int j = i + 1; j < NCELL; ++j) {
                if (!keep[j] || ct[ord[j]] != ci) continue;
                if (iou_xyxy(bi, &box[ord[j] * 4]) > 0.8f) keep[j] = 0;
            }
        }
        for (int r = 0; r < NCELL; ++r)
            out[(b * NCELL + ord[r]) * 7 + 6] = keep[r] ? 1.f : 0.f;
    }
}

// ---------------------------------------------------------------------------
// Host orchestration
// ---------------------------------------------------------------------------
static const int CFGV[21] = {64, 64, -1, 128, 128, -1, 256, 256, 256, 256, -1,
                             512, 512, 512, 512, -1, 512, 512, 512, 512, -1};

static void launch_conv(int Cin, int Cout, int H, const _Float16* cur,
                        const _Float16* wbuf, const float* bias, _Float16* nxt,
                        hipStream_t stream) {
    dim3 grid((H * H + 127) / 128, Cout / 64, BATCH);
#define CONV_CASE(ci, co, hh)                                                  \
    if (Cin == ci && Cout == co && H == hh) {                                  \
        conv3x3_wmma_t<ci, co, hh, hh><<<grid, 256, 0, stream>>>(              \
            cur, wbuf, bias, nxt);                                             \
        return;                                                                \
    }
    CONV_CASE(64, 64, 448)
    CONV_CASE(64, 128, 224)
    CONV_CASE(128, 128, 224)
    CONV_CASE(128, 256, 112)
    CONV_CASE(256, 256, 112)
    CONV_CASE(256, 512, 56)
    CONV_CASE(512, 512, 56)
    CONV_CASE(512, 512, 28)
#undef CONV_CASE
}

extern "C" void kernel_launch(void* const* d_in, const int* in_sizes, int n_in,
                              void* d_out, int out_size, void* d_ws, size_t ws_size,
                              hipStream_t stream) {
    (void)in_sizes; (void)n_in; (void)out_size; (void)ws_size;
    const float* imgs = (const float*)d_in[0];
    auto Wp = [&](int i) { return (const float*)d_in[3 + 2 * i]; };
    auto Bp = [&](int i) { return (const float*)d_in[4 + 2 * i]; };

    const size_t IMG_ELEMS = (size_t)BATCH * 3 * IMGSZ * IMGSZ;
    const size_t ACT_ELEMS = (size_t)BATCH * 64 * IMGSZ * IMGSZ;
    const size_t WB_ELEMS  = (size_t)512 * 512 * 9;

    char* base = (char*)d_ws;
    size_t off = 0;
    auto carve = [&](size_t bytes) -> char* {
        size_t s = (off + 255) & ~(size_t)255;
        off = s + bytes;
        return base + s;
    };
    _Float16* xh   = (_Float16*)carve(IMG_ELEMS * 2);
    _Float16* bufA = (_Float16*)carve(ACT_ELEMS * 2);
    _Float16* bufB = (_Float16*)carve(ACT_ELEMS * 2);
    _Float16* wbuf = (_Float16*)carve(WB_ELEMS * 2);
    float*    feat = (float*)carve((size_t)BATCH * COUT_FINAL * NCELL * 4);

    const int TH = 256;
    f32_to_f16_kernel<<<(int)((IMG_ELEMS + TH - 1) / TH), TH, 0, stream>>>(
        imgs, xh, (int)IMG_ELEMS);

    // layer 0: 3 -> 64 direct
    f32_to_f16_kernel<<<(64 * 27 + TH - 1) / TH, TH, 0, stream>>>(Wp(0), wbuf, 64 * 27);
    conv3x3_c3_kernel<<<(int)((BATCH * 64 * (size_t)IMGSZ * IMGSZ + TH - 1) / TH),
                        TH, 0, stream>>>(xh, wbuf, Bp(0), bufA);

    int C = 64, H = IMGSZ, pi = 1;
    _Float16* cur = bufA;
    _Float16* nxt = bufB;
    for (int ci = 1; ci < 21; ++ci) {
        int v = CFGV[ci];
        if (v < 0) {
            int tot = BATCH * C * (H / 2) * (H / 2);
            maxpool2x2_kernel<<<(tot + TH - 1) / TH, TH, 0, stream>>>(cur, nxt, C, H, H);
            H /= 2;
            _Float16* t = cur; cur = nxt; nxt = t;
        } else {
            int nw = v * C * 9;
            pack_w_kernel<<<(nw + TH - 1) / TH, TH, 0, stream>>>(Wp(pi), wbuf, v, C);
            launch_conv(C, v, H, cur, wbuf, Bp(pi), nxt, stream);
            _Float16* t = cur; cur = nxt; nxt = t;
            C = v; pi++;
        }
    }

    // final 1x1 head: 512 -> 30
    f32_to_f16_kernel<<<(COUT_FINAL * 512 + TH - 1) / TH, TH, 0, stream>>>(
        Wp(16), wbuf, COUT_FINAL * 512);
    conv1x1_wmma_kernel<<<dim3((NCELL + 15) / 16, 2, BATCH), 32, 0, stream>>>(
        cur, wbuf, Bp(16), feat);

    evaluate_kernel<<<BATCH, 224, 0, stream>>>(feat, (float*)d_out);
    nms_kernel<<<BATCH, 64, 0, stream>>>((float*)d_out);
}
